// PAR_73529840107623
// MI455X (gfx1250) — compile-verified, matrix operand order checked by reading, not run on hardware
//
#include <hip/hip_runtime.h>
#include <hip/hip_bf16.h>
#include <math.h>

// ---------------------------------------------------------------------------
// VAR transformer forward for MI455X (gfx1250, wave32, WMMA).
// All large matmuls use v_wmma_f32_16x16x32_bf16 (hardware f32->bf16 operand
// convert, f32 accumulate). Attention is a flash-style kernel with K/V^T
// staged in LDS (dynamic LDS ~113KB, legal on CDNA5's 320KB WGP LDS).
// ---------------------------------------------------------------------------

typedef __attribute__((ext_vector_type(16))) __bf16 v16bf;
typedef __attribute__((ext_vector_type(8)))  float  v8f;

union Frag { v16bf v; unsigned u[8]; };

// hardware bf16 convert
__device__ __forceinline__ unsigned short f2bfu(float f) {
  union { __hip_bfloat16 h; unsigned short s; } c;
  c.h = __float2bfloat16(f);
  return c.s;
}
__device__ __forceinline__ unsigned pack2f(float lo, float hi) {
  return (unsigned)f2bfu(lo) | ((unsigned)f2bfu(hi) << 16);
}

// level index of a sequence position (SCALES prefix sums)
__device__ __forceinline__ int lvl_of(int pos) {
  const int offs[13] = {1,3,6,10,16,25,38,57,84,124,184,272,400};
  int lv = 0;
#pragma unroll
  for (int i = 0; i < 13; ++i) lv += (pos >= offs[i]) ? 1 : 0;
  return lv;
}

// fragment K-pair base for 16-bit WMMA operand layout:
// element e (0..15) of lane (half = lane>>4) maps to K = e + (e>=8?8:0) + 8*half
__device__ __forceinline__ int kpair(int p, int half) {
  return 2 * p + 8 * half + ((p >= 4) ? 8 : 0);
}

// ---------------------------------------------------------------------------
// Generic GEMM: Out[M,N] = epi(A[M,K] @ W[K,N] + bias[N])
// EPI 0: none   EPI 1: exact GELU
// EPI 2: Out = Res + (AGate[col] + GGate[b*gstride+col]) * v , b = row/Lrows
// Block tile 128x128, 8 waves (2x4), each wave 64x32 (8 wmma / k-step).
// Requires: N % 128 == 0, K % 32 == 0. M guarded.
// ---------------------------------------------------------------------------
template <int EPI>
__global__ __launch_bounds__(256) void gemm_kernel(
    const float* __restrict__ A, const float* __restrict__ W,
    const float* __restrict__ bias, float* __restrict__ Out,
    int M, int N, int K,
    const float* __restrict__ Res,
    const float* __restrict__ AGate, const float* __restrict__ GGate,
    int gstride, int Lrows) {
  // packed bf16 pairs along K: stride 18 dwords (= 36 halves, conflict-free)
  __shared__ __align__(16) unsigned sA[128][18];   // [m][kpair]
  __shared__ __align__(16) unsigned sB[128][18];   // [n][kpair] (W transposed)

  const int tid  = threadIdx.x;
  const int lane = tid & 31, wid = tid >> 5;
  const int wm = wid >> 2, wn = wid & 3;           // 2 x 4 wave grid
  const int half = lane >> 4, l16 = lane & 15;
  const int m0 = blockIdx.y * 128;
  const int n0 = blockIdx.x * 128;

  v8f acc[4][2] = {};

  for (int k0 = 0; k0 < K; k0 += 32) {
    // A tile 128x32: float2 loads, packed bf16x2 stores (8 dwords / thread)
#pragma unroll
    for (int i = 0; i < 8; ++i) {
      int idx = tid + 256 * i;            // r*16 + kp
      int r = idx >> 4, kp = idx & 15;
      int gr = m0 + r;
      float2 v = make_float2(0.f, 0.f);
      if (gr < M) v = *(const float2*)&A[(size_t)gr * K + k0 + 2 * kp];
      sA[r][kp] = pack2f(v.x, v.y);
    }
    // W tile 32x128 -> transposed packed (two coalesced rows per dword)
#pragma unroll
    for (int i = 0; i < 8; ++i) {
      int idx = tid + 256 * i;            // kp*128 + n
      int nn = idx & 127, kp = idx >> 7;
      float w0 = W[(size_t)(k0 + 2 * kp) * N + n0 + nn];
      float w1 = W[(size_t)(k0 + 2 * kp + 1) * N + n0 + nn];
      sB[nn][kp] = pack2f(w0, w1);
    }
    __syncthreads();

    Frag fa[4], fb[2];
#pragma unroll
    for (int t = 0; t < 4; ++t) {
      int mr = wm * 64 + t * 16 + l16;
#pragma unroll
      for (int p = 0; p < 8; ++p) fa[t].u[p] = sA[mr][kpair(p, half) >> 1];
    }
#pragma unroll
    for (int j = 0; j < 2; ++j) {
      int nr = wn * 32 + j * 16 + l16;
#pragma unroll
      for (int p = 0; p < 8; ++p) fb[j].u[p] = sB[nr][kpair(p, half) >> 1];
    }
#pragma unroll
    for (int t = 0; t < 4; ++t)
#pragma unroll
      for (int j = 0; j < 2; ++j)
        acc[t][j] = __builtin_amdgcn_wmma_f32_16x16x32_bf16(
            false, fa[t].v, false, fb[j].v, (short)0, acc[t][j], false, false);
    __syncthreads();
  }

  // epilogue: C-layout lane holds col = l16, rows r + 8*half
#pragma unroll
  for (int t = 0; t < 4; ++t) {
#pragma unroll
    for (int j = 0; j < 2; ++j) {
      int col = n0 + wn * 32 + j * 16 + l16;
#pragma unroll
      for (int r = 0; r < 8; ++r) {
        int row = m0 + wm * 64 + t * 16 + r + 8 * half;
        if (row < M) {
          float v = acc[t][j][r] + bias[col];
          if (EPI == 1) v = 0.5f * v * (1.f + erff(v * 0.70710678118f));
          if (EPI == 2) {
            int b = row / Lrows;
            float g = AGate[col] + GGate[(size_t)b * gstride + col];
            v = Res[(size_t)row * N + col] + g * v;
          }
          Out[(size_t)row * N + col] = v;
        }
      }
    }
  }
}

// ---------------------------------------------------------------------------
// Flash attention per (b,h). grid.x = B*H. 8 waves, each owns 16 query rows.
// K stored [key][33] dwords (bf16x2 along d), V^T stored [d][209] dwords
// (bf16x2 along key). Level-causal mask: allowed iff lvl(key) <= lvl(query).
// ---------------------------------------------------------------------------
__global__ __launch_bounds__(256) void attn_kernel(
    const float* __restrict__ Q, const float* __restrict__ KV,
    float* __restrict__ O) {
  extern __shared__ unsigned smemu[];
  unsigned* Klu = smemu;                     // 400*33 dwords
  unsigned* Vtu = smemu + 400 * 33;          // 64*209 dwords
  unsigned short* Pl = (unsigned short*)(Vtu + 64 * 209);  // 8*16*36 halves

  const int bh = blockIdx.x;
  const int b = bh >> 4, h = bh & 15;
  const int tid = threadIdx.x, lane = tid & 31, wid = tid >> 5;
  const int half = lane >> 4, l16 = lane & 15;
  const size_t kv0 = (size_t)b * 400 * 2048;

  // K: pack pairs along d (contiguous in KV)
  for (int idx = tid; idx < 400 * 32; idx += 256) {
    int l = idx >> 5, dp = idx & 31;
    float2 kf = *(const float2*)&KV[kv0 + (size_t)l * 2048 + h * 64 + 2 * dp];
    Klu[l * 33 + dp] = pack2f(kf.x, kf.y);
  }
  // V^T: pack pairs along key
  for (int idx = tid; idx < 64 * 200; idx += 256) {
    int d = idx / 200, lp = idx % 200;
    float v0 = KV[kv0 + (size_t)(2 * lp)     * 2048 + 1024 + h * 64 + d];
    float v1 = KV[kv0 + (size_t)(2 * lp + 1) * 2048 + 1024 + h * 64 + d];
    Vtu[d * 209 + lp] = pack2f(v0, v1);
  }
  for (int idx = tid; idx < 64 * 9; idx += 256) {   // zero key tail 400..417
    int d = idx / 9, c = 200 + idx % 9;
    Vtu[d * 209 + c] = 0;
  }
  __syncthreads();

  unsigned short* Pw = Pl + wid * 16 * 36;

  for (int qt = wid; qt < 25; qt += 8) {
    int qb = qt * 16;
    // Q A-fragments (16x64 = two 16x32 frags), pre-scaled by 1/sqrt(DH)
    Frag aq[2];
    const size_t qrow = ((size_t)(b * 400 + qb + l16)) * 1024 + h * 64;
#pragma unroll
    for (int f = 0; f < 2; ++f)
#pragma unroll
      for (int p = 0; p < 8; ++p) {
        int kb = kpair(p, half) + 32 * f;
        float2 qq = *(const float2*)&Q[qrow + kb];
        aq[f].u[p] = pack2f(qq.x * 0.125f, qq.y * 0.125f);
      }
    int qlv[8];
#pragma unroll
    for (int r = 0; r < 8; ++r) qlv[r] = lvl_of(qb + r + 8 * half);

    float mrun[8], lrun[8];
#pragma unroll
    for (int r = 0; r < 8; ++r) { mrun[r] = -1e30f; lrun[r] = 0.f; }
    v8f oacc[4] = {};

    for (int kt = 0; kt < 13; ++kt) {        // 32 keys per step
      int kb0 = kt * 32;
      v8f s[2] = {};
#pragma unroll
      for (int t = 0; t < 2; ++t) {
        Frag bk[2];
        int key = kb0 + t * 16 + l16;
        bool valid = key < 400;
#pragma unroll
        for (int f = 0; f < 2; ++f)
#pragma unroll
          for (int p = 0; p < 8; ++p) {
            int kb = kpair(p, half) + 32 * f;
            bk[f].u[p] = valid ? Klu[key * 33 + (kb >> 1)] : 0u;
          }
        s[t] = __builtin_amdgcn_wmma_f32_16x16x32_bf16(
            false, aq[0].v, false, bk[0].v, (short)0, s[t], false, false);
        s[t] = __builtin_amdgcn_wmma_f32_16x16x32_bf16(
            false, aq[1].v, false, bk[1].v, (short)0, s[t], false, false);
      }
      int key0 = kb0 + l16, key1 = kb0 + 16 + l16;
      int klv0 = (key0 < 400) ? lvl_of(key0) : 99;
      int klv1 = (key1 < 400) ? lvl_of(key1) : 99;

      float p0[8], p1[8], corr[8];
#pragma unroll
      for (int r = 0; r < 8; ++r) {
        float s0 = (klv0 <= qlv[r]) ? s[0][r] : -1e30f;
        float s1 = (klv1 <= qlv[r]) ? s[1][r] : -1e30f;
        float mt = fmaxf(s0, s1);
#pragma unroll
        for (int off = 1; off < 16; off <<= 1)
          mt = fmaxf(mt, __shfl_xor(mt, off, 32));
        float mn = fmaxf(mrun[r], mt);
        float e0 = __expf(s0 - mn), e1 = __expf(s1 - mn);
        float rs = e0 + e1;
#pragma unroll
        for (int off = 1; off < 16; off <<= 1)
          rs += __shfl_xor(rs, off, 32);
        float cf = __expf(mrun[r] - mn);
        lrun[r] = lrun[r] * cf + rs;
        mrun[r] = mn;
        corr[r] = cf;
        p0[r] = e0; p1[r] = e1;
      }
#pragma unroll
      for (int t2 = 0; t2 < 4; ++t2)
#pragma unroll
        for (int r = 0; r < 8; ++r) oacc[t2][r] *= corr[r];

      // C-layout P -> LDS -> A-layout fragment (wave-private scratch)
#pragma unroll
      for (int r = 0; r < 8; ++r) {
        int row = r + 8 * half;
        Pw[row * 36 + l16]      = f2bfu(p0[r]);
        Pw[row * 36 + 16 + l16] = f2bfu(p1[r]);
      }
      Frag ap;
#pragma unroll
      for (int p = 0; p < 8; ++p)
        ap.u[p] = *(const unsigned*)&Pw[l16 * 36 + kpair(p, half)];

#pragma unroll
      for (int t2 = 0; t2 < 4; ++t2) {
        Frag bv;
        int d = t2 * 16 + l16;
#pragma unroll
        for (int p = 0; p < 8; ++p)
          bv.u[p] = Vtu[d * 209 + ((kb0 + kpair(p, half)) >> 1)];
        oacc[t2] = __builtin_amdgcn_wmma_f32_16x16x32_bf16(
            false, ap.v, false, bv.v, (short)0, oacc[t2], false, false);
      }
    }
    // normalize + store
#pragma unroll
    for (int r = 0; r < 8; ++r) {
      float inv = 1.f / lrun[r];
      int row = qb + r + 8 * half;
#pragma unroll
      for (int t2 = 0; t2 < 4; ++t2)
        O[((size_t)(b * 400 + row)) * 1024 + h * 64 + t2 * 16 + l16] =
            oacc[t2][r] * inv;
    }
  }
}

// ---------------------------------------------------------------------------
// LayerNorm with ada modulation: out = ln(x)*(1 + a_s[c] + g_s[b,c]) + (a_h[c] + g_h[b,c])
// One block per row (C=1024, 256 threads x 4).
// ---------------------------------------------------------------------------
__global__ __launch_bounds__(256) void ln_mod_kernel(
    const float* __restrict__ X, float* __restrict__ Out,
    const float* __restrict__ AScale, const float* __restrict__ GScale,
    const float* __restrict__ AShift, const float* __restrict__ GShift,
    int gstride, int Lrows) {
  const int row = blockIdx.x;
  const int b = row / Lrows;
  const int tid = threadIdx.x;
  __shared__ float red[8];

  float v[4];
  float s = 0.f;
#pragma unroll
  for (int i = 0; i < 4; ++i) {
    v[i] = X[(size_t)row * 1024 + tid + 256 * i];
    s += v[i];
  }
#pragma unroll
  for (int off = 16; off >= 1; off >>= 1) s += __shfl_xor(s, off, 32);
  if ((tid & 31) == 0) red[tid >> 5] = s;
  __syncthreads();
  float mean = 0.f;
#pragma unroll
  for (int i = 0; i < 8; ++i) mean += red[i];
  mean *= (1.f / 1024.f);
  __syncthreads();

  float vs = 0.f;
#pragma unroll
  for (int i = 0; i < 4; ++i) { float d = v[i] - mean; vs += d * d; }
#pragma unroll
  for (int off = 16; off >= 1; off >>= 1) vs += __shfl_xor(vs, off, 32);
  if ((tid & 31) == 0) red[tid >> 5] = vs;
  __syncthreads();
  float var = 0.f;
#pragma unroll
  for (int i = 0; i < 8; ++i) var += red[i];
  var *= (1.f / 1024.f);
  float rstd = rsqrtf(var + 1e-6f);

#pragma unroll
  for (int i = 0; i < 4; ++i) {
    int c = tid + 256 * i;
    float sc = (AScale ? AScale[c] : 0.f) + GScale[(size_t)b * gstride + c];
    float sh = (AShift ? AShift[c] : 0.f) + GShift[(size_t)b * gstride + c];
    Out[(size_t)row * 1024 + c] = (v[i] - mean) * rstd * (1.f + sc) + sh;
  }
}

// ---------------------------------------------------------------------------
// Small elementwise kernels
// ---------------------------------------------------------------------------
__global__ void silu_kernel(const float* __restrict__ in, float* __restrict__ out, int n) {
  int i = blockIdx.x * 256 + threadIdx.x;
  if (i < n) { float v = in[i]; out[i] = v / (1.f + __expf(-v)); }
}

__global__ void lvl_kernel(const float* __restrict__ lvl_tab,
                           const float* __restrict__ pos, float* __restrict__ lvl) {
  int i = blockIdx.x * 256 + threadIdx.x;
  if (i < 400 * 1024) {
    int l = i >> 10;
    lvl[i] = lvl_tab[lvl_of(l) * 1024 + (i & 1023)] + pos[i];
  }
}

__global__ void assemble_x_kernel(const float* __restrict__ cond,
                                  const float* __restrict__ pos_start,
                                  const float* __restrict__ emb,
                                  const float* __restrict__ lvl,
                                  float* __restrict__ x) {
  int i = blockIdx.x * 256 + threadIdx.x;
  if (i < 32 * 400 * 1024) {
    int c = i & 1023;
    int rl = i >> 10;
    int l = rl % 400, b = rl / 400;
    float base = (l == 0) ? (cond[b * 1024 + c] + pos_start[c])
                          : emb[((size_t)(b * 399 + l - 1)) * 1024 + c];
    x[i] = base + lvl[l * 1024 + c];
  }
}

__global__ void add_lvl_kernel(float* __restrict__ poc, const float* __restrict__ lvl) {
  int i = blockIdx.x * 256 + threadIdx.x;
  if (i < 32 * 400 * 1024) {
    int l = (i >> 10) % 400;
    poc[i] += lvl[(size_t)l * 1024 + (i & 1023)];
  }
}

// ---------------------------------------------------------------------------
// Host orchestration
// ---------------------------------------------------------------------------
extern "C" void kernel_launch(void* const* d_in, const int* in_sizes, int n_in,
                              void* d_out, int out_size, void* d_ws, size_t ws_size,
                              hipStream_t stream) {
  const float* x_BLCv     = (const float*)d_in[0];
  const float* poc_ctx    = (const float*)d_in[1];
  const float* poc_cond   = (const float*)d_in[2];
  const float* word_w     = (const float*)d_in[3];
  const float* word_b     = (const float*)d_in[4];
  const float* poc_w      = (const float*)d_in[5];
  const float* poc_b      = (const float*)d_in[6];
  const float* pos_start  = (const float*)d_in[7];
  const float* pos_1LC    = (const float*)d_in[8];
  const float* lvl_tab    = (const float*)d_in[9];
  const float* ada_w      = (const float*)d_in[10];
  const float* ada_b      = (const float*)d_in[11];
  const float* ada_gss    = (const float*)d_in[12];
  const float* q_w        = (const float*)d_in[13];
  const float* q_b        = (const float*)d_in[14];
  const float* kv_w       = (const float*)d_in[15];
  const float* kv_b       = (const float*)d_in[16];
  const float* proj_w     = (const float*)d_in[17];
  const float* proj_b     = (const float*)d_in[18];
  const float* fc1_w      = (const float*)d_in[19];
  const float* fc1_b      = (const float*)d_in[20];
  const float* fc2_w      = (const float*)d_in[21];
  const float* fc2_b      = (const float*)d_in[22];
  const float* head_ada_w = (const float*)d_in[23];
  const float* head_ada_b = (const float*)d_in[24];
  const float* head_w     = (const float*)d_in[25];
  const float* head_b     = (const float*)d_in[26];

  const int B = 32, L = 400, C = 1024, MLP = 4096, V = 4096, ND = 256, CVAE = 64;
  const int BL = B * L;                        // 12800

  float* ws = (float*)d_ws;
  size_t off = 0;
  auto alloc = [&](size_t n) { float* p = ws + off; off += n; return p; };
  float* lvl    = alloc((size_t)L * C);
  float* cond   = alloc((size_t)B * C);
  float* scond  = alloc((size_t)B * C);
  float* gss    = alloc((size_t)B * 6 * C);
  float* hss    = alloc((size_t)B * 2 * C);
  float* x      = alloc((size_t)BL * C);
  float* poc    = alloc((size_t)BL * C);
  float* xn     = alloc((size_t)BL * C);
  float* q      = alloc((size_t)BL * C);       // also embedding scratch
  float* kv     = alloc((size_t)BL * 2 * C);
  float* o      = alloc((size_t)BL * C);
  float* hidden = alloc((size_t)BL * MLP);
  (void)ws_size; (void)n_in; (void)in_sizes; (void)out_size;

  auto gemm = [&](int epi, const float* A, const float* W, const float* bias,
                  float* Out, int M, int N, int K, const float* Res,
                  const float* AG, const float* GG, int gstride) {
    dim3 grid(N / 128, (M + 127) / 128);
    if (epi == 0)
      gemm_kernel<0><<<grid, 256, 0, stream>>>(A, W, bias, Out, M, N, K,
                                               nullptr, nullptr, nullptr, 0, L);
    else if (epi == 1)
      gemm_kernel<1><<<grid, 256, 0, stream>>>(A, W, bias, Out, M, N, K,
                                               nullptr, nullptr, nullptr, 0, L);
    else
      gemm_kernel<2><<<grid, 256, 0, stream>>>(A, W, bias, Out, M, N, K,
                                               Res, AG, GG, gstride, L);
  };

  const int EW = 256;
  lvl_kernel<<<(L * C + EW - 1) / EW, EW, 0, stream>>>(lvl_tab, pos_1LC, lvl);

  // cond = poc_cond @ poc_w + poc_b ; scond = silu(cond)
  gemm(0, poc_cond, poc_w, poc_b, cond, B, C, ND, nullptr, nullptr, nullptr, 0);
  silu_kernel<<<(B * C + EW - 1) / EW, EW, 0, stream>>>(cond, scond, B * C);

  // gss / head ss
  gemm(0, scond, ada_w, ada_b, gss, B, 6 * C, C, nullptr, nullptr, nullptr, 0);
  gemm(0, scond, head_ada_w, head_ada_b, hss, B, 2 * C, C, nullptr, nullptr, nullptr, 0);

  // token embedding + x assembly
  gemm(0, x_BLCv, word_w, word_b, q /*tmp*/, B * (L - 1), C, CVAE,
       nullptr, nullptr, nullptr, 0);
  assemble_x_kernel<<<(BL * C + EW - 1) / EW, EW, 0, stream>>>(cond, pos_start, q, lvl, x);

  // poc context embedding
  gemm(0, poc_ctx, word_w, word_b, poc, BL, C, CVAE, nullptr, nullptr, nullptr, 0);
  add_lvl_kernel<<<(BL * C + EW - 1) / EW, EW, 0, stream>>>(poc, lvl);

  const size_t attn_lds = (size_t)(400 * 33 + 64 * 209 + 8 * 16 * 18) * sizeof(unsigned);

  for (int d = 0; d < 8; ++d) {
    const float* ag = ada_gss + (size_t)d * 6 * C;   // [6][C] rows: g1,g2,s1,s2,sh1,sh2
    // ln + (s1, sh1)
    ln_mod_kernel<<<BL, 256, 0, stream>>>(x, xn, ag + 2 * C, gss + 2 * C,
                                          ag + 4 * C, gss + 4 * C, 6 * C, L);
    gemm(0, xn, q_w + (size_t)d * C * C, q_b + (size_t)d * C, q, BL, C, C,
         nullptr, nullptr, nullptr, 0);
    gemm(0, poc, kv_w + (size_t)d * C * 2 * C, kv_b + (size_t)d * 2 * C, kv,
         BL, 2 * C, C, nullptr, nullptr, nullptr, 0);
    attn_kernel<<<B * 16, 256, attn_lds, stream>>>(q, kv, o);
    // x += g1 * (o @ proj_w + proj_b)
    gemm(2, o, proj_w + (size_t)d * C * C, proj_b + (size_t)d * C, x, BL, C, C,
         x, ag + 0, gss + 0, 6 * C);
    // ln + (s2, sh2)
    ln_mod_kernel<<<BL, 256, 0, stream>>>(x, xn, ag + 3 * C, gss + 3 * C,
                                          ag + 5 * C, gss + 5 * C, 6 * C, L);
    gemm(1, xn, fc1_w + (size_t)d * C * MLP, fc1_b + (size_t)d * MLP, hidden,
         BL, MLP, C, nullptr, nullptr, nullptr, 0);
    // x += g2 * (gelu_hidden @ fc2_w + fc2_b)
    gemm(2, hidden, fc2_w + (size_t)d * MLP * C, fc2_b + (size_t)d * C, x, BL, C, MLP,
         x, ag + 1 * C, gss + 1 * C, 6 * C);
  }

  // head: ln with (1 + hss[:, :C]) and hss[:, C:], then logits
  ln_mod_kernel<<<BL, 256, 0, stream>>>(x, xn, nullptr, hss, nullptr, hss + C, 2 * C, L);
  gemm(0, xn, head_w, head_b, (float*)d_out, BL, V, C, nullptr, nullptr, nullptr, 0);
}
